// Attention_6545530159375
// MI455X (gfx1250) — compile-verified, matrix operand order checked by reading, not run on hardware
//
#include <hip/hip_runtime.h>
#include <hip/hip_bf16.h>

// ---------------------------------------------------------------------------
// LLaMA-style attention layer for gfx1250 (MI455X), bf16 WMMA everywhere.
// B=2, S=2048, D=4096, H=32, KV=8, HD=128, causal, RoPE theta=5e5.
// ---------------------------------------------------------------------------

typedef __attribute__((ext_vector_type(16))) __bf16 v16bf;
typedef __attribute__((ext_vector_type(8)))  __bf16 v8bf;
typedef __attribute__((ext_vector_type(8)))  float  v8f;

#define B_   2
#define S_   2048
#define D_   4096
#define H_   32
#define KV_  8
#define HD_  128
#define NREP_ 4

// LDS row stride (elements) for staged tiles: 40 bf16 = 80 B = 20 words.
// Row r -> bank word (20*r) mod 64: 16 consecutive rows hit 16 distinct,
// 4-word-spaced offsets => conflict-free b128 fragment reads.
#define LDP 40

// ---------------------------------------------------------------------------
// Fragment loader (works for A 16x32 MxK and B 32x16 KxN when the "row"
// direction is K-contiguous). ISA 7.12.2 16-bit layout:
// lane<16: row=lane, K = koff..koff+7, koff+16..koff+23, koff=(lane>>4)*8.
// ---------------------------------------------------------------------------
__device__ __forceinline__ v16bf load_frag(const __bf16* base, int ld) {
    int lane = threadIdx.x & 31;
    int row  = lane & 15;
    int koff = (lane >> 4) << 3;
    const __bf16* p = base + (size_t)row * (size_t)ld + koff;
    v8bf lo = *(const v8bf*)(p);
    v8bf hi = *(const v8bf*)(p + 16);
    v16bf r;
#pragma unroll
    for (int i = 0; i < 8; ++i) { r[i] = lo[i]; r[i + 8] = hi[i]; }
    return r;
}

// ---------------------------------------------------------------------------
// fp32 -> bf16 elementwise convert
// ---------------------------------------------------------------------------
__global__ void cvt_f32_to_bf16(const float* __restrict__ src,
                                __bf16* __restrict__ dst, int n) {
    int i = blockIdx.x * blockDim.x + threadIdx.x;
    if (i < n) dst[i] = (__bf16)src[i];
}

// ---------------------------------------------------------------------------
// GEMM: Out[M,N] = A[M,K] * W[N,K]^T  (both bf16 row-major, K contiguous)
// Block = 256 threads = 8 waves. Block tile 128M x 128N, K-step 32, staged
// through LDS. Waves in 2(M) x 4(N) grid; each wave: 64M x 32N = 4x2 WMMA
// tiles, 8 WMMA per K-step. Global traffic 16KB per 1.05 MFLOP K-step.
// ---------------------------------------------------------------------------
template <bool OUT_BF16>
__global__ __launch_bounds__(256) void gemm_bf16_nt(
    const __bf16* __restrict__ A, const __bf16* __restrict__ W,
    void* __restrict__ Out, int M, int N, int K) {
    __shared__ __bf16 a_s[128 * LDP];
    __shared__ __bf16 w_s[128 * LDP];

    int t    = threadIdx.x;
    int wave = t >> 5;
    int lane = t & 31;
    int bm0  = blockIdx.y * 128;
    int bn0  = blockIdx.x * 128;
    int wr   = wave >> 2;          // 0..1  -> M offset wr*64
    int wc   = wave & 3;           // 0..3  -> N offset wc*32

    // staging assignment: each thread moves one 16-element half-row of A and W
    int srow  = t >> 1;            // 0..127
    int shalf = (t & 1) << 4;      // 0 or 16
    const __bf16* ga = A + (size_t)(bm0 + srow) * K + shalf;
    const __bf16* gw = W + (size_t)(bn0 + srow) * K + shalf;
    __bf16* la = &a_s[srow * LDP + shalf];
    __bf16* lw = &w_s[srow * LDP + shalf];

    v8f acc[4][2];
#pragma unroll
    for (int i = 0; i < 4; ++i)
#pragma unroll
        for (int j = 0; j < 2; ++j) acc[i][j] = (v8f){};

    for (int kk = 0; kk < K; kk += 32) {
        __syncthreads();           // previous iteration's reads done
        *(v8bf*)(la)     = *(const v8bf*)(ga + kk);
        *(v8bf*)(la + 8) = *(const v8bf*)(ga + kk + 8);
        *(v8bf*)(lw)     = *(const v8bf*)(gw + kk);
        *(v8bf*)(lw + 8) = *(const v8bf*)(gw + kk + 8);
        __syncthreads();           // tile visible to all waves

        v16bf b0 = load_frag(&w_s[(wc * 32) * LDP], LDP);
        v16bf b1 = load_frag(&w_s[(wc * 32 + 16) * LDP], LDP);
#pragma unroll
        for (int i = 0; i < 4; ++i) {
            v16bf a = load_frag(&a_s[(wr * 64 + i * 16) * LDP], LDP);
            acc[i][0] = __builtin_amdgcn_wmma_f32_16x16x32_bf16(
                false, a, false, b0, (short)0, acc[i][0], false, false);
            acc[i][1] = __builtin_amdgcn_wmma_f32_16x16x32_bf16(
                false, a, false, b1, (short)0, acc[i][1], false, false);
        }
    }

    // C/D layout: vgpr r, lanes 0-15 -> (M=r, N=lane); lanes 16-31 -> (M=8+r)
    int nloc  = lane & 15;
    int half8 = (lane >> 4) << 3;
#pragma unroll
    for (int i = 0; i < 4; ++i)
#pragma unroll
        for (int j = 0; j < 2; ++j)
#pragma unroll
            for (int r = 0; r < 8; ++r) {
                size_t m = (size_t)(bm0 + wr * 64 + i * 16 + half8 + r);
                size_t n = (size_t)(bn0 + wc * 32 + j * 16 + nloc);
                if (OUT_BF16) ((__bf16*)Out)[m * N + n] = (__bf16)acc[i][j][r];
                else          ((float*)Out)[m * N + n]  = acc[i][j][r];
            }
}

// ---------------------------------------------------------------------------
// RoPE in place on bf16 tensor [b][s][nh][hd]; interleaved pairs rotated by
// (cos[s][i], sin[s][i]). One thread per pair. total = B*S*nh*(HD/2).
// ---------------------------------------------------------------------------
__global__ void rope_bf16(__bf16* __restrict__ t, const float* __restrict__ cosb,
                          const float* __restrict__ sinb, int nh, int total) {
    int i = blockIdx.x * blockDim.x + threadIdx.x;
    if (i >= total) return;
    int p     = i & (HD_ / 2 - 1);           // 0..63
    int rowid = i >> 6;                      // (b*S + s)*nh + h
    int s     = (rowid / nh) % S_;
    size_t base = (size_t)rowid * HD_ + 2 * p;
    float tr = (float)t[base];
    float ti = (float)t[base + 1];
    float c  = cosb[s * (HD_ / 2) + p];
    float sn = sinb[s * (HD_ / 2) + p];
    t[base]     = (__bf16)(tr * c - ti * sn);
    t[base + 1] = (__bf16)(tr * sn + ti * c);
}

// ---------------------------------------------------------------------------
// Transpose V: [b][s][kv][hd] -> [b][kv][hd][s]  (bf16)
// ---------------------------------------------------------------------------
__global__ void v_transpose(const __bf16* __restrict__ v, __bf16* __restrict__ vt,
                            int total) {
    int i = blockIdx.x * blockDim.x + threadIdx.x;
    if (i >= total) return;
    int hd = i & (HD_ - 1);
    int kv = (i >> 7) & (KV_ - 1);
    int s  = (i >> 10) & (S_ - 1);
    int b  = i >> 21;
    vt[(((size_t)b * KV_ + kv) * HD_ + hd) * S_ + s] = v[i];
}

// ---------------------------------------------------------------------------
// Flash attention, causal. One wave per (b, h, 16-query tile).
// q:  [b][s][H][HD]   (bf16, roped)
// k:  [b][s][KV][HD]  (bf16, roped)
// vt: [b][kv][HD][S]  (bf16)
// o:  [b][s][H][HD]   (bf16)
// ---------------------------------------------------------------------------
__global__ __launch_bounds__(256) void flash_attn(
    const __bf16* __restrict__ q, const __bf16* __restrict__ k,
    const __bf16* __restrict__ vt, __bf16* __restrict__ o) {
    __shared__ __bf16 pshm[8 * 16 * 32];     // per-wave 16x32 P staging
    int wave = threadIdx.x >> 5;
    int lane = threadIdx.x & 31;
    int tile = blockIdx.x * 8 + wave;        // 0 .. B*H*(S/16)-1
    int qt  = tile & (S_ / 16 - 1);
    int h   = (tile >> 7) & (H_ - 1);
    int b   = tile >> 12;
    int kvh = h / NREP_;
    int q0  = qt * 16;
    __bf16* pb = pshm + wave * (16 * 32);

    // resident Q fragments over HD = 4 x 32
    const __bf16* qbase = q + (((size_t)(b * S_ + q0)) * H_ + h) * HD_;
    v16bf qf[4];
#pragma unroll
    for (int kt = 0; kt < 4; ++kt) qf[kt] = load_frag(qbase + kt * 32, H_ * HD_);

    v8f acc[8];
#pragma unroll
    for (int tt = 0; tt < 8; ++tt) acc[tt] = (v8f){};
    float mrow[8], lrow[8];
#pragma unroll
    for (int r = 0; r < 8; ++r) { mrow[r] = -3.0e38f; lrow[r] = 0.0f; }

    int nloc  = lane & 15;
    int half8 = (lane >> 4) << 3;
    const float scale = 0.08838834764831845f;   // 1/sqrt(128)

    for (int k0 = 0; k0 < q0 + 16; k0 += 32) {
        // ---- scores: two 16x16 tiles over 32 keys ----
        v8f sc[2];
#pragma unroll
        for (int j = 0; j < 2; ++j) {
            const __bf16* kbase =
                k + (((size_t)(b * S_ + k0 + j * 16)) * KV_ + kvh) * HD_;
            v8f c = {};
#pragma unroll
            for (int kt = 0; kt < 4; ++kt) {
                v16bf kf = load_frag(kbase + kt * 32, KV_ * HD_);
                c = __builtin_amdgcn_wmma_f32_16x16x32_bf16(false, qf[kt], false, kf,
                                                            (short)0, c, false, false);
            }
            sc[j] = c;
        }
        // ---- scale + causal mask ----
#pragma unroll
        for (int j = 0; j < 2; ++j) {
            int ncol = k0 + j * 16 + nloc;
#pragma unroll
            for (int r = 0; r < 8; ++r) {
                int mr = q0 + half8 + r;
                float vvv = sc[j][r] * scale;
                sc[j][r] = (ncol > mr) ? -1.0e9f : vvv;
            }
        }
        // ---- online softmax (rows live per half-wave; xor-reduce 16 lanes) ----
        float alpha[8];
#pragma unroll
        for (int r = 0; r < 8; ++r) {
            float mx = fmaxf(sc[0][r], sc[1][r]);
            for (int x = 1; x < 16; x <<= 1) mx = fmaxf(mx, __shfl_xor(mx, x, 32));
            float mnew = fmaxf(mrow[r], mx);
            alpha[r] = __expf(mrow[r] - mnew);
            float p0 = __expf(sc[0][r] - mnew);
            float p1 = __expf(sc[1][r] - mnew);
            sc[0][r] = p0; sc[1][r] = p1;
            float rs = p0 + p1;
            for (int x = 1; x < 16; x <<= 1) rs += __shfl_xor(rs, x, 32);
            lrow[r] = lrow[r] * alpha[r] + rs;
            mrow[r] = mnew;
        }
#pragma unroll
        for (int tt = 0; tt < 8; ++tt)
#pragma unroll
            for (int r = 0; r < 8; ++r) acc[tt][r] *= alpha[r];

        // ---- transpose P (C layout -> A layout) via LDS ----
#pragma unroll
        for (int j = 0; j < 2; ++j)
#pragma unroll
            for (int r = 0; r < 8; ++r)
                pb[(half8 + r) * 32 + j * 16 + nloc] = (__bf16)sc[j][r];
        asm volatile("s_wait_dscnt 0" ::: "memory");
        v16bf pf = load_frag(pb, 32);

        // ---- O += P * V : 8 hd-tiles of 16 ----
#pragma unroll
        for (int tt = 0; tt < 8; ++tt) {
            const __bf16* vbase =
                vt + (((size_t)b * KV_ + kvh) * HD_ + tt * 16) * S_ + k0;
            v16bf vf = load_frag(vbase, S_);
            acc[tt] = __builtin_amdgcn_wmma_f32_16x16x32_bf16(false, pf, false, vf,
                                                              (short)0, acc[tt], false, false);
        }
    }

    // ---- epilogue: O /= l, store [b][s][H][HD] bf16 ----
#pragma unroll
    for (int r = 0; r < 8; ++r) lrow[r] = 1.0f / lrow[r];
#pragma unroll
    for (int tt = 0; tt < 8; ++tt)
#pragma unroll
        for (int r = 0; r < 8; ++r) {
            size_t idx = (((size_t)(b * S_ + q0 + half8 + r)) * H_ + h) * HD_ +
                         tt * 16 + nloc;
            o[idx] = (__bf16)(acc[tt][r] * lrow[r]);
        }
}

// ---------------------------------------------------------------------------
// Host-side launch.
// ---------------------------------------------------------------------------
extern "C" void kernel_launch(void* const* d_in, const int* in_sizes, int n_in,
                              void* d_out, int out_size, void* d_ws, size_t ws_size,
                              hipStream_t stream) {
    (void)in_sizes; (void)n_in; (void)out_size; (void)ws_size;
    const float* x    = (const float*)d_in[0];
    const float* wq   = (const float*)d_in[1];
    const float* wk   = (const float*)d_in[2];
    const float* wv   = (const float*)d_in[3];
    const float* wo   = (const float*)d_in[4];
    const float* cosb = (const float*)d_in[5];
    const float* sinb = (const float*)d_in[6];
    // d_in[7] (mask) and d_in[8] (start_pos=0) are folded into the inline causal mask.

    const size_t NX   = (size_t)B_ * S_ * D_;        // 16777216
    const size_t NWQ  = (size_t)H_ * HD_ * D_;       // 16777216
    const size_t NWK  = (size_t)KV_ * HD_ * D_;      // 4194304
    const size_t NQ   = (size_t)B_ * S_ * H_ * HD_;  // 16777216
    const size_t NK   = (size_t)B_ * S_ * KV_ * HD_; // 4194304

    char* ws = (char*)d_ws;
    size_t off = 0;
    __bf16* xb  = (__bf16*)(ws + off); off += NX  * 2;
    __bf16* wqb = (__bf16*)(ws + off); off += NWQ * 2;
    __bf16* wkb = (__bf16*)(ws + off); off += NWK * 2;
    __bf16* wvb = (__bf16*)(ws + off); off += NWK * 2;
    __bf16* wob = (__bf16*)(ws + off); off += NWQ * 2;
    __bf16* qb  = (__bf16*)(ws + off); off += NQ  * 2;
    __bf16* kb  = (__bf16*)(ws + off); off += NK  * 2;
    __bf16* vb  = (__bf16*)(ws + off); off += NK  * 2;
    __bf16* vtb = (__bf16*)(ws + off); off += NK  * 2;
    __bf16* ab  = (__bf16*)(ws + off); off += NQ  * 2;   // attention output

    // 1) fp32 -> bf16 conversions
    cvt_f32_to_bf16<<<(int)((NX  + 255) / 256), 256, 0, stream>>>(x,  xb,  (int)NX);
    cvt_f32_to_bf16<<<(int)((NWQ + 255) / 256), 256, 0, stream>>>(wq, wqb, (int)NWQ);
    cvt_f32_to_bf16<<<(int)((NWK + 255) / 256), 256, 0, stream>>>(wk, wkb, (int)NWK);
    cvt_f32_to_bf16<<<(int)((NWK + 255) / 256), 256, 0, stream>>>(wv, wvb, (int)NWK);
    cvt_f32_to_bf16<<<(int)((NWQ + 255) / 256), 256, 0, stream>>>(wo, wob, (int)NWQ);

    const int M = B_ * S_;        // 4096
    // 2) QKV projections (A @ W^T), bf16 out; block tile 128x128
    gemm_bf16_nt<true><<<dim3((H_ * HD_) / 128, M / 128), 256, 0, stream>>>(
        xb, wqb, qb, M, H_ * HD_, D_);
    gemm_bf16_nt<true><<<dim3((KV_ * HD_) / 128, M / 128), 256, 0, stream>>>(
        xb, wkb, kb, M, KV_ * HD_, D_);
    gemm_bf16_nt<true><<<dim3((KV_ * HD_) / 128, M / 128), 256, 0, stream>>>(
        xb, wvb, vb, M, KV_ * HD_, D_);

    // 3) RoPE in place on Q and K
    {
        int totq = B_ * S_ * H_ * (HD_ / 2);
        int totk = B_ * S_ * KV_ * (HD_ / 2);
        rope_bf16<<<(totq + 255) / 256, 256, 0, stream>>>(qb, cosb, sinb, H_, totq);
        rope_bf16<<<(totk + 255) / 256, 256, 0, stream>>>(kb, cosb, sinb, KV_, totk);
    }

    // 4) V transpose to [b][kv][hd][s]
    {
        int totv = B_ * S_ * KV_ * HD_;
        v_transpose<<<(totv + 255) / 256, 256, 0, stream>>>(vb, vtb, totv);
    }

    // 5) causal flash attention: B*H*(S/16) = 8192 wave-tiles, 8 per block
    flash_attn<<<(B_ * H_ * (S_ / 16)) / 8, 256, 0, stream>>>(qb, kb, vtb, ab);

    // 6) output projection, fp32 out
    gemm_bf16_nt<false><<<dim3(D_ / 128, M / 128), 256, 0, stream>>>(
        ab, wob, (float*)d_out, M, D_, H_ * HD_);
}
